// FusedNormMMReplacement_1889785610793
// MI455X (gfx1250) — compile-verified, matrix operand order checked by reading, not run on hardware
//
#include <hip/hip_runtime.h>
#include <hip/hip_bf16.h>

typedef __attribute__((ext_vector_type(16))) __bf16   v16bf;
typedef __attribute__((ext_vector_type(8)))  float    v8f;
typedef __attribute__((ext_vector_type(4)))  float    v4f;
typedef __attribute__((ext_vector_type(4)))  unsigned v4u;
typedef __attribute__((ext_vector_type(8)))  int      v8i;
typedef __attribute__((ext_vector_type(4)))  int      v4i;

#define HDIM 2048
#define BM 128
#define BN 128
#define BK 64
#define ROWB 128   // unpadded LDS row bytes (fallback path, XOR swizzled)
#define ROWP 144   // TDM-padded LDS row bytes (128B data + 16B pad)

struct Frag32B { uint4 lo, hi; };

__device__ __forceinline__ unsigned pack_bf16(float a, float b) {
  unsigned ua = __builtin_bit_cast(unsigned, a);
  unsigned ub = __builtin_bit_cast(unsigned, b);
  ua += 0x7FFFu + ((ua >> 16) & 1u);
  ub += 0x7FFFu + ((ub >> 16) & 1u);
  return (ua >> 16) | (ub & 0xFFFF0000u);
}

__device__ __forceinline__ int sw_off(int row, int byteCol) {
  int chunk  = byteCol >> 4;
  int within = byteCol & 15;
  return row * ROWB + (((chunk ^ (row & 7)) << 4) | within);
}

// ---------------- LayerNorm -> bf16 ----------------
__global__ __launch_bounds__(256) void ln_bf16_kernel(
    const float* __restrict__ x, const float* __restrict__ gamma,
    const float* __restrict__ beta, unsigned short* __restrict__ normed) {
  const int row = blockIdx.x;
  const int t   = threadIdx.x;
  const v4f* xr = reinterpret_cast<const v4f*>(x + (size_t)row * HDIM);
  // input is a read-once 64MB stream -> non-temporal loads keep L2 for tiles
  v4f v0 = __builtin_nontemporal_load(&xr[2 * t]);
  v4f v1 = __builtin_nontemporal_load(&xr[2 * t + 1]);

  float s  = v0.x + v0.y + v0.z + v0.w + v1.x + v1.y + v1.z + v1.w;
  float ss = v0.x * v0.x + v0.y * v0.y + v0.z * v0.z + v0.w * v0.w +
             v1.x * v1.x + v1.y * v1.y + v1.z * v1.z + v1.w * v1.w;

  __shared__ float r0[256], r1[256];
  r0[t] = s; r1[t] = ss;
  __syncthreads();
  for (int off = 128; off > 0; off >>= 1) {
    if (t < off) { r0[t] += r0[t + off]; r1[t] += r1[t + off]; }
    __syncthreads();
  }
  const float mu  = r0[0] * (1.0f / HDIM);
  const float var = r1[0] * (1.0f / HDIM) - mu * mu;
  const float rs  = rsqrtf(var + 1e-5f);

  const float4* g = reinterpret_cast<const float4*>(gamma);
  const float4* b = reinterpret_cast<const float4*>(beta);
  float4 g0 = g[2 * t], g1 = g[2 * t + 1];
  float4 b0 = b[2 * t], b1 = b[2 * t + 1];

  float y0 = (v0.x - mu) * rs * g0.x + b0.x;
  float y1 = (v0.y - mu) * rs * g0.y + b0.y;
  float y2 = (v0.z - mu) * rs * g0.z + b0.z;
  float y3 = (v0.w - mu) * rs * g0.w + b0.w;
  float y4 = (v1.x - mu) * rs * g1.x + b1.x;
  float y5 = (v1.y - mu) * rs * g1.y + b1.y;
  float y6 = (v1.z - mu) * rs * g1.z + b1.z;
  float y7 = (v1.w - mu) * rs * g1.w + b1.w;

  uint4 o;
  o.x = pack_bf16(y0, y1); o.y = pack_bf16(y2, y3);
  o.z = pack_bf16(y4, y5); o.w = pack_bf16(y6, y7);
  reinterpret_cast<uint4*>(normed + (size_t)row * HDIM)[t] = o;
}

// ---------------- fp32 -> bf16 weight conversion ----------------
__global__ __launch_bounds__(256) void cvt_bf16_kernel(
    const float* __restrict__ src, unsigned short* __restrict__ dst) {
  const int i = blockIdx.x * 256 + threadIdx.x;  // one float4 per thread
  const v4f v = __builtin_nontemporal_load(&reinterpret_cast<const v4f*>(src)[i]);
  uint2 o;
  o.x = pack_bf16(v.x, v.y);
  o.y = pack_bf16(v.z, v.w);
  reinterpret_cast<uint2*>(dst)[i] = o;  // re-read by GEMM: keep temporal
}

// ---------------- TDM descriptor build + issue ----------------
// 2-D tile: tile_dim0 = 64 bf16 per row, tile_dim1 = 64 rows,
// tensor_dim0 = 2048, row stride 2048; LDS pad 16B per 128B row -> 144B stride.
__device__ __forceinline__ void tdm_load_tile(unsigned lds_off, const void* gsrc,
                                              unsigned tensor_rows) {
  const unsigned long long ga = (unsigned long long)(uintptr_t)gsrc;
  v4u g0;
  g0.x = 1u;                                           // count = 1 valid D#
  g0.y = lds_off;                                      // lds_addr
  g0.z = (unsigned)ga;                                 // global_addr[31:0]
  g0.w = ((unsigned)(ga >> 32) & 0x01ffffffu) | (2u << 30);  // ga[56:32] | type=2
  v8i g1;
  g1[0] = (int)((1u << 16)      // data_size = 2 bytes
              | (1u << 20)      // pad_enable
              | (4u << 22)      // pad_interval: 32 DWORDs (= one 128B row)
              | (3u << 25));    // pad_amount: 4 DWORDs (= 16B)
  g1[1] = (int)(2048u << 16);   // tensor_dim0[15:0] in d1[31:16]
  g1[2] = (int)((tensor_rows & 0xffffu) << 16);  // tensor_dim1[15:0] in d2[31:16]
  g1[3] = (int)(64u << 16);     // tile_dim0 = 64 in d3[31:16]
  g1[4] = 64;                   // tile_dim1 = 64 rows
  g1[5] = 2048;                 // tensor_dim0_stride (elements)
  g1[6] = 0;
  g1[7] = 0;
  v4i zz = (v4i)0;
#if defined(__clang_major__) && (__clang_major__ >= 23)
  v8i zz8 = (v8i)0;
  __builtin_amdgcn_tensor_load_to_lds(g0, g1, zz, zz, zz8, 0);
#else
  __builtin_amdgcn_tensor_load_to_lds(g0, g1, zz, zz, 0);
#endif
}

// ---------------- Fused QKV GEMM, TDM-fed (preferred path) ----------------
__global__ __launch_bounds__(256) void qkv_wmma_tdm_kernel(
    const unsigned short* __restrict__ normed,
    const unsigned short* __restrict__ wbf,  // [3][H][H] bf16
    const float* __restrict__ bq, const float* __restrict__ bk,
    const float* __restrict__ bv, float* __restrict__ out, int Mrows) {
  __shared__ __align__(16) char smemA[2][BM * ROWP];
  __shared__ __align__(16) char smemB[2][BN * ROWP];

  const int z = blockIdx.z;
  const float* bias = (z == 0) ? bq : (z == 1) ? bk : bv;
  const unsigned short* W = wbf + (size_t)z * HDIM * HDIM;

  const int m0 = blockIdx.x * BM;
  const int n0 = blockIdx.y * BN;
  const int t = threadIdx.x;
  const int lane = t & 31, wid = t >> 5;
  const int half = lane >> 4, lm = lane & 15;
  const int wm = wid & 1, wn = wid >> 1;

  // LDS byte offsets of the tile buffers (generic ptr low 32 bits = LDS offset)
  const unsigned ldsA0 = (unsigned)(uintptr_t)&smemA[0][0];
  const unsigned ldsA1 = (unsigned)(uintptr_t)&smemA[1][0];
  const unsigned ldsB0 = (unsigned)(uintptr_t)&smemB[0][0];
  const unsigned ldsB1 = (unsigned)(uintptr_t)&smemB[1][0];

  v8f acc[4][2];
#pragma unroll
  for (int i = 0; i < 4; ++i)
#pragma unroll
    for (int j = 0; j < 2; ++j) acc[i][j] = (v8f)(0.0f);

  constexpr int NS = HDIM / BK;  // 32 K-stages

  // waves 0,1: A tile halves; waves 2,3: B tile halves (each 64 rows x 64 cols)
  auto issue = [&](int s, int buf) {
    if (wid < 4) {
      const int k0 = s * BK;
      const int hw = wid & 1;
      if (wid < 2) {
        const unsigned dst = (buf ? ldsA1 : ldsA0) + (unsigned)(hw * 64 * ROWP);
        tdm_load_tile(dst, normed + (size_t)(m0 + hw * 64) * HDIM + k0,
                      (unsigned)Mrows);
      } else {
        const unsigned dst = (buf ? ldsB1 : ldsB0) + (unsigned)(hw * 64 * ROWP);
        tdm_load_tile(dst, W + (size_t)(n0 + hw * 64) * HDIM + k0, HDIM);
      }
    }
  };

  auto compute = [&](int buf) {
    const char* aB = smemA[buf];
    const char* bB = smemB[buf];
#pragma unroll
    for (int kk = 0; kk < BK; kk += 32) {
      v16bf aF[4], bF[2];
#pragma unroll
      for (int mt = 0; mt < 4; ++mt) {
        const char* p = aB + (wm * 64 + mt * 16 + lm) * ROWP + kk * 2 + half * 16;
        Frag32B f;
        f.lo = *reinterpret_cast<const uint4*>(p);
        f.hi = *reinterpret_cast<const uint4*>(p + 32);
        aF[mt] = __builtin_bit_cast(v16bf, f);
      }
#pragma unroll
      for (int nt = 0; nt < 2; ++nt) {
        const char* p = bB + (wn * 32 + nt * 16 + lm) * ROWP + kk * 2 + half * 16;
        Frag32B f;
        f.lo = *reinterpret_cast<const uint4*>(p);
        f.hi = *reinterpret_cast<const uint4*>(p + 32);
        bF[nt] = __builtin_bit_cast(v16bf, f);
      }
#pragma unroll
      for (int mt = 0; mt < 4; ++mt)
#pragma unroll
        for (int nt = 0; nt < 2; ++nt)
          acc[mt][nt] = __builtin_amdgcn_wmma_f32_16x16x32_bf16(
              false, aF[mt], false, bF[nt], (short)0, acc[mt][nt], false, false);
    }
  };

  issue(0, 0);
  __builtin_amdgcn_s_wait_tensorcnt(0);
  __syncthreads();
  for (int s = 0; s < NS; ++s) {
    if (s + 1 < NS) issue(s + 1, (s + 1) & 1);
    compute(s & 1);
    __builtin_amdgcn_s_wait_tensorcnt(0);  // issuer waits its DMA; no-op for others
    __syncthreads();
  }

  // write-once 192MB stream: non-temporal stores, don't pollute L2
  const int zoff = z * HDIM;
#pragma unroll
  for (int mt = 0; mt < 4; ++mt) {
#pragma unroll
    for (int nt = 0; nt < 2; ++nt) {
      const int go = n0 + wn * 32 + nt * 16 + lm;
      const float bv_ = bias[go];
      const int mbase = m0 + wm * 64 + mt * 16 + half * 8;
#pragma unroll
      for (int r = 0; r < 8; ++r) {
        __builtin_nontemporal_store(
            acc[mt][nt][r] + bv_,
            &out[(size_t)(mbase + r) * (3 * HDIM) + zoff + go]);
      }
    }
  }
}

// ---------------- Fallback GEMM (register staging, fp32 W on the fly) -------
__global__ __launch_bounds__(256) void qkv_wmma_kernel(
    const unsigned short* __restrict__ normed,
    const float* __restrict__ wq, const float* __restrict__ wk,
    const float* __restrict__ wv,
    const float* __restrict__ bq, const float* __restrict__ bk,
    const float* __restrict__ bv, float* __restrict__ out) {
  __shared__ __align__(16) char smem[2][2][BM * ROWB];

  const int z = blockIdx.z;
  const float* W    = (z == 0) ? wq : (z == 1) ? wk : wv;
  const float* bias = (z == 0) ? bq : (z == 1) ? bk : bv;

  const int m0 = blockIdx.x * BM;
  const int n0 = blockIdx.y * BN;
  const int t    = threadIdx.x;
  const int lane = t & 31, wid = t >> 5;
  const int half = lane >> 4, lm = lane & 15;
  const int wm = wid & 1;
  const int wn = wid >> 1;

  v8f acc[4][2];
#pragma unroll
  for (int i = 0; i < 4; ++i)
#pragma unroll
    for (int j = 0; j < 2; ++j) acc[i][j] = (v8f)(0.0f);

  uint4 aR[4];
  uint2 bR[8];
  constexpr int NS = HDIM / BK;

  auto gload = [&](int s) {
    const int k0 = s * BK;
#pragma unroll
    for (int i = 0; i < 4; ++i) {
      const int c = t + 256 * i;
      const int row = c >> 3, cc = c & 7;
      aR[i] = *reinterpret_cast<const uint4*>(
          normed + (size_t)(m0 + row) * HDIM + k0 + cc * 8);
    }
#pragma unroll
    for (int i = 0; i < 8; ++i) {
      const int c = t + 256 * i;
      const int row = c >> 4, cc = c & 15;
      const float4 w4 = *reinterpret_cast<const float4*>(
          W + (size_t)(n0 + row) * HDIM + k0 + cc * 4);
      bR[i].x = pack_bf16(w4.x, w4.y);
      bR[i].y = pack_bf16(w4.z, w4.w);
    }
  };

  auto sstore = [&](int buf) {
    char* aB = smem[buf][0];
    char* bB = smem[buf][1];
#pragma unroll
    for (int i = 0; i < 4; ++i) {
      const int c = t + 256 * i;
      const int row = c >> 3, cc = c & 7;
      *reinterpret_cast<uint4*>(aB + sw_off(row, cc * 16)) = aR[i];
    }
#pragma unroll
    for (int i = 0; i < 8; ++i) {
      const int c = t + 256 * i;
      const int row = c >> 4, cc = c & 15;
      *reinterpret_cast<uint2*>(bB + sw_off(row, cc * 8)) = bR[i];
    }
  };

  auto compute = [&](int buf) {
    const char* aB = smem[buf][0];
    const char* bB = smem[buf][1];
#pragma unroll
    for (int kk = 0; kk < BK; kk += 32) {
      v16bf aF[4], bF[2];
#pragma unroll
      for (int mt = 0; mt < 4; ++mt) {
        const int row = wm * 64 + mt * 16 + lm;
        const int col = kk * 2 + half * 16;
        Frag32B f;
        f.lo = *reinterpret_cast<const uint4*>(aB + sw_off(row, col));
        f.hi = *reinterpret_cast<const uint4*>(aB + sw_off(row, col + 32));
        aF[mt] = __builtin_bit_cast(v16bf, f);
      }
#pragma unroll
      for (int nt = 0; nt < 2; ++nt) {
        const int row = wn * 32 + nt * 16 + lm;
        const int col = kk * 2 + half * 16;
        Frag32B f;
        f.lo = *reinterpret_cast<const uint4*>(bB + sw_off(row, col));
        f.hi = *reinterpret_cast<const uint4*>(bB + sw_off(row, col + 32));
        bF[nt] = __builtin_bit_cast(v16bf, f);
      }
#pragma unroll
      for (int mt = 0; mt < 4; ++mt)
#pragma unroll
        for (int nt = 0; nt < 2; ++nt)
          acc[mt][nt] = __builtin_amdgcn_wmma_f32_16x16x32_bf16(
              false, aF[mt], false, bF[nt], (short)0, acc[mt][nt], false, false);
    }
  };

  gload(0);
  sstore(0);
  __syncthreads();
  for (int s = 0; s < NS; ++s) {
    if (s + 1 < NS) gload(s + 1);
    if (s + 2 < NS) {
      const int k2 = (s + 2) * BK;
      __builtin_prefetch(normed + (size_t)(m0 + (t >> 3)) * HDIM + k2 + (t & 7) * 8, 0, 0);
      __builtin_prefetch(W + (size_t)(n0 + (t >> 4)) * HDIM + k2 + (t & 15) * 4, 0, 0);
    }
    compute(s & 1);
    if (s + 1 < NS) {
      sstore((s + 1) & 1);
      __syncthreads();
    }
  }

  const int zoff = z * HDIM;
#pragma unroll
  for (int mt = 0; mt < 4; ++mt) {
#pragma unroll
    for (int nt = 0; nt < 2; ++nt) {
      const int go = n0 + wn * 32 + nt * 16 + lm;
      const float bv_ = bias[go];
      const int mbase = m0 + wm * 64 + mt * 16 + half * 8;
#pragma unroll
      for (int r = 0; r < 8; ++r) {
        __builtin_nontemporal_store(
            acc[mt][nt][r] + bv_,
            &out[(size_t)(mbase + r) * (3 * HDIM) + zoff + go]);
      }
    }
  }
}

extern "C" void kernel_launch(void* const* d_in, const int* in_sizes, int n_in,
                              void* d_out, int out_size, void* d_ws, size_t ws_size,
                              hipStream_t stream) {
  const float* x     = (const float*)d_in[0];
  const float* gamma = (const float*)d_in[1];
  const float* beta  = (const float*)d_in[2];
  const float* wq    = (const float*)d_in[3];
  const float* bq    = (const float*)d_in[4];
  const float* wk    = (const float*)d_in[5];
  const float* bk    = (const float*)d_in[6];
  const float* wv    = (const float*)d_in[7];
  const float* bv    = (const float*)d_in[8];
  float* out = (float*)d_out;

  const int M = in_sizes[0] / HDIM;  // B*S = 8192
  const size_t normedB = (size_t)M * HDIM * 2;        // 32 MB
  const size_t wB      = 3ull * HDIM * HDIM * 2;      // 24 MB

  unsigned short* normed = (unsigned short*)d_ws;
  ln_bf16_kernel<<<M, 256, 0, stream>>>(x, gamma, beta, normed);

  dim3 grid(M / BM, HDIM / BN, 3);
  if (ws_size >= normedB + wB) {
    unsigned short* wbf = (unsigned short*)((char*)d_ws + normedB);
    const int nblk = (HDIM * HDIM / 4) / 256;  // float4 per thread
    cvt_bf16_kernel<<<nblk, 256, 0, stream>>>(wq, wbf);
    cvt_bf16_kernel<<<nblk, 256, 0, stream>>>(wk, wbf + (size_t)HDIM * HDIM);
    cvt_bf16_kernel<<<nblk, 256, 0, stream>>>(wv, wbf + 2ull * HDIM * HDIM);
    qkv_wmma_tdm_kernel<<<grid, 256, 0, stream>>>(normed, wbf, bq, bk, bv, out, M);
  } else {
    qkv_wmma_kernel<<<grid, 256, 0, stream>>>(normed, wq, wk, wv, bq, bk, bv, out);
  }
}
